// AttentionRetrievalHead_47605417508997
// MI455X (gfx1250) — matrix operand order repaired and checked by reading, not compiled, not run on hardware
//
#include <hip/hip_runtime.h>

// ---------------------------------------------------------------------------
// AttentionRetrievalHead for MI455X (gfx1250, wave32, WMMA)
//
// B=64, N=262144, DM=256, H=8, HD=32, K=64
//
// Pipeline:
//  1) qk_project : qkh[b,h,m] = sum_d q[b,h*32+d]*Wk[h*32+d,m]  (f32, ws)
//                  qk_bf16[b,m] = sum_h qkh                      (bf16, ws)
//  2) score_wmma : avg[b,n] = (scale/H) * qk[b]·memory[n]  via
//                  v_wmma_f32_16x16x32_bf16, masked with -inf.   (f32, ws)
//  3) topk       : exact per-batch top-64 via 4-round radix select.
//  4) finalize   : per-head scores for selected rows (from qkh),
//                  softmax, weighted memory rows -> Wv proj (+bv),
//                  output proj Wo (+bo), attn_weights, idx-as-float.
// ---------------------------------------------------------------------------

typedef __attribute__((ext_vector_type(16))) __bf16 v16bf;
typedef __attribute__((ext_vector_type(8)))  __bf16 v8bf;
typedef __attribute__((ext_vector_type(8)))  float  v8f;

namespace {
constexpr int kB  = 64;
constexpr int kN  = 262144;
constexpr int kDM = 256;
constexpr int kH  = 8;
constexpr int kHD = 32;
constexpr int kK  = 64;
constexpr float kScale    = 0.17677669529663687f;   // HD^-0.5 = 1/sqrt(32)
constexpr float kScaleAvg = 0.022097086912079608f;  // kScale / H
}

// ---------------------------------------------------------------------------
// Phase 1: tiny per-head projection of the queries through Wk.
// qkh[(b*8+h)*256+m], qkbf[b*256+m]. 16384 threads, ~4 MFLOP total.
// ---------------------------------------------------------------------------
__global__ __launch_bounds__(256) void qk_project_kernel(
    const float* __restrict__ query, const float* __restrict__ Wk,
    float* __restrict__ qkh, __bf16* __restrict__ qkbf)
{
    const int id = blockIdx.x * 256 + threadIdx.x;   // 0..16383
    const int b = id >> 8;
    const int m = id & 255;
    float tot = 0.0f;
    for (int h = 0; h < kH; ++h) {
        float s = 0.0f;
        #pragma unroll 8
        for (int d = 0; d < kHD; ++d) {
            const int j = h * kHD + d;
            s += query[b * kDM + j] * Wk[j * kDM + m];
        }
        qkh[(b * kH + h) * kDM + m] = s;
        tot += s;
    }
    qkbf[b * kDM + m] = (__bf16)tot;
}

// ---------------------------------------------------------------------------
// Phase 2: avg[b,n] for all n via WMMA.  One wave owns 16 memory rows (WMMA N
// dim); the 64 queries are 4 M-tiles reusing the B fragment; K=256 in 8 steps
// of 32.  Block = 8 waves = 128 rows; grid = N/128 = 2048 blocks.
// ---------------------------------------------------------------------------
__global__ __launch_bounds__(256) void score_wmma_kernel(
    const float* __restrict__ memory, const __bf16* __restrict__ qkbf,
    const unsigned char* __restrict__ mask, float* __restrict__ avg)
{
    __shared__ __align__(16) __bf16 qk_lds[kB * kDM];   // 32 KB

    const int tid = threadIdx.x;
    // Cooperative copy of qk (bf16) into LDS: 32 KB, 8B per thread-iter.
    {
        const unsigned long long* src = (const unsigned long long*)qkbf;
        unsigned long long*       dst = (unsigned long long*)qk_lds;
        for (int i = tid; i < kB * kDM / 4; i += 256) dst[i] = src[i];
    }
    __syncthreads();

    const int lane  = tid & 31;
    const int wave  = tid >> 5;
    const int half  = lane >> 4;      // 0: lanes 0-15, 1: lanes 16-31
    const int l16   = lane & 15;
    const int nbase = (blockIdx.x * 8 + wave) * 16;
    const int n     = nbase + l16;    // this lane's memory row

    v8f acc[4];
    #pragma unroll
    for (int t = 0; t < 4; ++t) acc[t] = v8f{};

    const float* mrow = memory + (size_t)n * kDM + half * 16;
    const int base8 = half * 8;       // A-layout K sub-offset per half-wave

    #pragma unroll
    for (int s = 0; s < 8; ++s) {
        const int k0 = s * 32;

        // B fragment (32x16 bf16, K x N): lane holds memory[n, k0+half*16 .. +15]
        const float4* mp = (const float4*)(mrow + k0);
        float4 f0 = mp[0], f1 = mp[1], f2 = mp[2], f3 = mp[3];
        v16bf bfrag;
        bfrag[0]  = (__bf16)f0.x; bfrag[1]  = (__bf16)f0.y;
        bfrag[2]  = (__bf16)f0.z; bfrag[3]  = (__bf16)f0.w;
        bfrag[4]  = (__bf16)f1.x; bfrag[5]  = (__bf16)f1.y;
        bfrag[6]  = (__bf16)f1.z; bfrag[7]  = (__bf16)f1.w;
        bfrag[8]  = (__bf16)f2.x; bfrag[9]  = (__bf16)f2.y;
        bfrag[10] = (__bf16)f2.z; bfrag[11] = (__bf16)f2.w;
        bfrag[12] = (__bf16)f3.x; bfrag[13] = (__bf16)f3.y;
        bfrag[14] = (__bf16)f3.z; bfrag[15] = (__bf16)f3.w;

        // 4 M-tiles of queries share this B fragment.
        #pragma unroll
        for (int t = 0; t < 4; ++t) {
            const int m = t * 16 + l16;
            union { v16bf v; v8bf h[2]; } ua;
            // 16-bit A layout (matches B): lanes 0-15 hold K {0..15},
            // lanes 16-31 hold K {16..31} of this 32-wide k-step.
            ua.h[0] = *(const v8bf*)&qk_lds[m * kDM + k0 + half * 16];
            ua.h[1] = *(const v8bf*)&qk_lds[m * kDM + k0 + half * 16 + 8];
            acc[t] = __builtin_amdgcn_wmma_f32_16x16x32_bf16(
                false, ua.v, false, bfrag, (short)0, acc[t], false, false);
        }
    }

    // C/D layout: element r of lane -> (M = t*16 + r + half*8, N = l16).
    const bool mk = mask[n] != 0;
    #pragma unroll
    for (int t = 0; t < 4; ++t) {
        #pragma unroll
        for (int r = 0; r < 8; ++r) {
            const int brow = t * 16 + r + half * 8;
            const float v = acc[t][r] * kScaleAvg;
            avg[(size_t)brow * kN + n] = mk ? v : -__builtin_inff();
        }
    }
}

// ---------------------------------------------------------------------------
// Phase 3: exact top-64 per batch row via 4-round radix select on the
// float->ordered-uint key (256-bin LDS histogram), then parallel rank sort.
// ---------------------------------------------------------------------------
__device__ __forceinline__ unsigned ordkey(float f) {
    unsigned u = __float_as_uint(f);
    return (u & 0x80000000u) ? ~u : (u | 0x80000000u);
}

__global__ __launch_bounds__(256) void topk_kernel(
    const float* __restrict__ avg, int* __restrict__ topidx)
{
    __shared__ unsigned hist[256];
    __shared__ unsigned s_prefix, s_remaining, s_cnt_hi, s_cnt_tie;
    __shared__ float sel_val[kK];
    __shared__ int   sel_idx[kK];
    __shared__ int   tie_idx[kK];

    const int b   = blockIdx.x;
    const int tid = threadIdx.x;
    const float* row = avg + (size_t)b * kN;

    if (tid == 0) { s_prefix = 0u; s_remaining = kK; }
    __syncthreads();

    unsigned prefix_mask = 0u;
    for (int round = 0; round < 4; ++round) {
        const int shift = 24 - 8 * round;
        hist[tid] = 0u;
        __syncthreads();
        const unsigned prefix = s_prefix;
        for (int n = tid; n < kN; n += 256) {
            const unsigned u = ordkey(row[n]);
            if ((u & prefix_mask) == prefix)
                atomicAdd(&hist[(u >> shift) & 255u], 1u);
        }
        __syncthreads();
        if (tid == 0) {
            unsigned rem = s_remaining;
            int bin = 255;
            for (; bin > 0; --bin) {
                if (hist[bin] >= rem) break;
                rem -= hist[bin];
            }
            s_remaining = rem;
            s_prefix = prefix | ((unsigned)bin << shift);
        }
        __syncthreads();
        prefix_mask |= (0xFFu << shift);
    }

    const unsigned T   = s_prefix;      // exact ordered-uint of the K-th value
    const unsigned rem = s_remaining;   // #ties at T to include (>=1)
    if (tid == 0) { s_cnt_hi = 0u; s_cnt_tie = 0u; }
    __syncthreads();

    for (int n = tid; n < kN; n += 256) {
        const float f = row[n];
        const unsigned u = ordkey(f);
        if (u > T) {
            const unsigned p = atomicAdd(&s_cnt_hi, 1u);   // exactly kK-rem total
            sel_val[p] = f; sel_idx[p] = n;
        } else if (u == T) {
            const unsigned p = atomicAdd(&s_cnt_tie, 1u);
            if (p < rem) tie_idx[p] = n;
        }
    }
    __syncthreads();

    const unsigned cnt_hi = s_cnt_hi;   // == kK - rem
    if (tid < (int)rem) {
        sel_idx[cnt_hi + tid] = tie_idx[tid];
        sel_val[cnt_hi + tid] =
            __uint_as_float((T >> 31) ? (T & 0x7FFFFFFFu) : ~T);
    }
    __syncthreads();

    // O(K^2) parallel rank sort: descending value, ties by ascending index.
    if (tid < kK) {
        const float v  = sel_val[tid];
        const int   ix = sel_idx[tid];
        int rank = 0;
        for (int j = 0; j < kK; ++j) {
            const float vj = sel_val[j];
            const int   ij = sel_idx[j];
            if (vj > v || (vj == v && ij < ix)) ++rank;
        }
        topidx[b * kK + rank] = ix;
    }
}

// ---------------------------------------------------------------------------
// Phase 4: tail — per-head scores on selected rows, softmax, weighted memory
// rows, Wv projection (+bv), Wo projection (+bo).  One block per batch.
// ---------------------------------------------------------------------------
__global__ __launch_bounds__(256) void finalize_kernel(
    const float* __restrict__ memory, const float* __restrict__ qkh,
    const float* __restrict__ Wv, const float* __restrict__ bv,
    const float* __restrict__ Wo, const float* __restrict__ bo,
    const int* __restrict__ topidx, float* __restrict__ out)
{
    __shared__ int   idx[kK];
    __shared__ float s[kH][kK];     // scores -> probs
    __shared__ float wm[kH][kDM];   // softmax-weighted memory rows
    __shared__ float ctx[kDM];      // context before Wo

    const int b   = blockIdx.x;
    const int tid = threadIdx.x;

    if (tid < kK) idx[tid] = topidx[b * kK + tid];
    __syncthreads();

    // per-head scores on the 64 selected rows (bk shifts cancel in softmax)
    for (int p = tid; p < kH * kK; p += 256) {
        const int h = p >> 6, k = p & 63;
        const float* mr = memory + (size_t)idx[k] * kDM;
        const float* qr = qkh + ((size_t)b * kH + h) * kDM;
        float acc = 0.0f;
        for (int m = 0; m < kDM; ++m) acc += qr[m] * mr[m];
        s[h][k] = acc * kScale;
    }
    __syncthreads();

    // softmax along k for each head (tiny; 8 threads)
    if (tid < kH) {
        const int h = tid;
        float mx = s[h][0];
        for (int k = 1; k < kK; ++k) mx = fmaxf(mx, s[h][k]);
        float sum = 0.0f;
        for (int k = 0; k < kK; ++k) { const float e = expf(s[h][k] - mx); s[h][k] = e; sum += e; }
        const float inv = 1.0f / sum;
        for (int k = 0; k < kK; ++k) s[h][k] *= inv;
    }
    __syncthreads();

    // attn_weights[b,k] = mean over heads
    if (tid < kK) {
        float a = 0.0f;
        for (int h = 0; h < kH; ++h) a += s[h][tid];
        out[kB * kDM + b * kK + tid] = a * 0.125f;
    }

    // wm[h][m] = sum_k p[h][k] * memory[idx_k][m]
    for (int p = tid; p < kH * kDM; p += 256) {
        const int h = p >> 8, m = p & 255;
        float acc = 0.0f;
        for (int k = 0; k < kK; ++k) acc += s[h][k] * memory[(size_t)idx[k] * kDM + m];
        wm[h][m] = acc;
    }
    __syncthreads();

    // ctx[j] = wm[j/32] . Wv[j,:] + bv[j]   (softmax weights sum to 1)
    {
        const int j = tid, h = j >> 5;
        float acc = bv[j];
        for (int m = 0; m < kDM; ++m) acc += wm[h][m] * Wv[j * kDM + m];
        ctx[j] = acc;
    }
    __syncthreads();

    // out_ctx[b,j] = ctx . Wo[j,:] + bo[j]
    {
        const int j = tid;
        float acc = bo[j];
        for (int i = 0; i < kDM; ++i) acc += ctx[i] * Wo[j * kDM + i];
        out[b * kDM + j] = acc;
    }

    // top_idx emitted as float (third output)
    if (tid < kK) out[kB * kDM + kB * kK + b * kK + tid] = (float)idx[tid];
}

// ---------------------------------------------------------------------------
// Workspace layout (bytes):
//   [0,            524288)  qkh   : 64*8*256 f32
//   [524288,       557056)  qkbf  : 64*256 bf16
//   [557056,     67665920)  avg   : 64*262144 f32 (64 MB)
//   [67665920,   67682304)  topidx: 64*64 i32
// ---------------------------------------------------------------------------
extern "C" void kernel_launch(void* const* d_in, const int* in_sizes, int n_in,
                              void* d_out, int out_size, void* d_ws, size_t ws_size,
                              hipStream_t stream) {
    const float*         query  = (const float*)d_in[0];
    const float*         memory = (const float*)d_in[1];
    const float*         Wk     = (const float*)d_in[2];
    // d_in[3] = bk: a constant shift per (b,h) in scores / per b in avg —
    // cancels exactly in both top-k and softmax, so it never affects outputs.
    const float*         Wv     = (const float*)d_in[4];
    const float*         bv     = (const float*)d_in[5];
    const float*         Wo     = (const float*)d_in[6];
    const float*         bo     = (const float*)d_in[7];
    const unsigned char* mask   = (const unsigned char*)d_in[8];
    // d_in[9] = top_k scalar (== 64, compile-time constant kK)
    (void)in_sizes; (void)n_in; (void)out_size; (void)ws_size;

    char* ws = (char*)d_ws;
    float*  qkh    = (float*)(ws);
    __bf16* qkbf   = (__bf16*)(ws + 524288);
    float*  avg    = (float*)(ws + 557056);
    int*    topidx = (int*)(ws + 67665920);
    float*  out    = (float*)d_out;

    qk_project_kernel<<<kB, 256, 0, stream>>>(query, Wk, qkh, qkbf);
    score_wmma_kernel<<<kN / 128, 256, 0, stream>>>(memory, qkbf, mask, avg);
    topk_kernel<<<kB, 256, 0, stream>>>(avg, topidx);
    finalize_kernel<<<kB, 256, 0, stream>>>(memory, qkh, Wv, bv, Wo, bo, topidx, out);
}
